// LNLL_emb_87479893885401
// MI455X (gfx1250) — compile-verified, matrix operand order checked by reading, not run on hardware
//
#include <hip/hip_runtime.h>
#include <math.h>

typedef __attribute__((ext_vector_type(16))) _Float16 v16h;
typedef __attribute__((ext_vector_type(8)))  float    v8f;

#define EPS_F    1.1920928955078125e-7f
#define LAMBDA_C 0.01f
#define W_POS    0.7f
#define W_NEG    0.3f
#define N_GROUPS 64
#define GROUP_W  16
#define CAT_COLS 1024
#define HIDDEN   1024
#define IN_FEATS 2048
#define BATCH    8388608

#define BCE_BLOCKS  2048
#define BCE_THREADS 256

// ---------------- Kernel 1: BCE partial sums (bandwidth-bound streaming) ----
__device__ __forceinline__ float nll_term(float p, float y) {
    float pc = fminf(fmaxf(p,        EPS_F), 1.0f - EPS_F);
    float qc = fminf(fmaxf(1.0f - p, EPS_F), 1.0f - EPS_F);
    float lp = __logf(pc);
    float lq = __logf(qc);
    return -(W_POS * y * lp + W_NEG * (1.0f - y) * lq);
}

__global__ void bce_partial_kernel(const float* __restrict__ p,
                                   const float* __restrict__ y,
                                   float* __restrict__ partial) {
    const float4* p4 = (const float4*)p;
    const float4* y4 = (const float4*)y;
    const int tid    = threadIdx.x;
    const int gid    = blockIdx.x * BCE_THREADS + tid;
    const int stride = BCE_BLOCKS * BCE_THREADS;
    const int N4     = BATCH / 4;

    float acc = 0.0f;
    for (int i = gid; i < N4; i += stride) {
        float4 pv = p4[i];
        float4 yv = y4[i];
        acc += nll_term(pv.x, yv.x);
        acc += nll_term(pv.y, yv.y);
        acc += nll_term(pv.z, yv.z);
        acc += nll_term(pv.w, yv.w);
    }

    __shared__ float sdata[BCE_THREADS];
    sdata[tid] = acc;
    __syncthreads();
    for (int s = BCE_THREADS / 2; s > 0; s >>= 1) {
        if (tid < s) sdata[tid] += sdata[tid + s];
        __syncthreads();
    }
    if (tid == 0) partial[blockIdx.x] = sdata[0];
}

// ---------------- Kernel 2: column L2 norms of W via WMMA ------------------
// col_sum[j] = sum_k W[k][j]^2 computed as D = A x B (+C) with
//   A[m][k] = W^2[kbase+k][c0+m]  (f16, 16x32 tile, ISA lane layout)
//   B       = all-ones f16        (so D[m][n] = sum_k A[m][k] for every n)
// Accumulated over HIDDEN/32 = 32 chunks in the f32 C operand.
__global__ void colnorm_wmma_kernel(const float* __restrict__ W,
                                    float* __restrict__ colnorm) {
    const int wave = (blockIdx.x * blockDim.x + threadIdx.x) >> 5;  // 0..127
    const int lane = threadIdx.x & 31;
    const int c0   = wave * 16;            // 16 columns per wave
    const int m    = lane & 15;            // A-matrix row index M
    const int col  = c0 + m;
    const int khalf = (lane < 16) ? 0 : 8; // K sub-block per half-wave

    v16h ones;
#pragma unroll
    for (int i = 0; i < 16; ++i) ones[i] = (_Float16)1.0f;

    v8f c = {};
    for (int kb = 0; kb < HIDDEN; kb += 32) {
        v16h a;
#pragma unroll
        for (int j = 0; j < 8; ++j) {
            // VGPR j holds K = khalf+2j,khalf+2j+1 (j<4) else 16+khalf+2(j-4)..+1
            int kk = (j < 4) ? (khalf + 2 * j) : (16 + khalf + 2 * (j - 4));
            float w0 = W[(size_t)(kb + kk)     * IN_FEATS + col];
            float w1 = W[(size_t)(kb + kk + 1) * IN_FEATS + col];
            a[2 * j]     = (_Float16)(w0 * w0);
            a[2 * j + 1] = (_Float16)(w1 * w1);
        }
        c = __builtin_amdgcn_wmma_f32_16x16x32_f16(
                /*neg_a=*/false, a, /*neg_b=*/false, ones,
                /*c_mod=*/(short)0, c, /*reuse_a=*/false, /*reuse_b=*/false);
    }

    // D layout (32-bit C/D 16x16): VGPR r -> lanes 0-15: (M=r, N=lane),
    // lanes 16-31: (M=8+r, N=lane-16). Take the N=0 column.
    if (lane == 0) {
#pragma unroll
        for (int r = 0; r < 8; ++r) colnorm[c0 + r] = sqrtf(c[r]);
    } else if (lane == 16) {
#pragma unroll
        for (int r = 0; r < 8; ++r) colnorm[c0 + 8 + r] = sqrtf(c[r]);
    }
}

// ---------------- Kernel 3: penalty + final combine ------------------------
__device__ __forceinline__ float penalty_f(float x) {
    return 1.0f - __expf(-(x * x)) + fabsf(x);   // beta = 1.0
}

__global__ void final_combine_kernel(const float* __restrict__ partial,
                                     const float* __restrict__ colnorm,
                                     float* __restrict__ out) {
    __shared__ float sb[256];
    __shared__ float sr[256];
    const int tid = threadIdx.x;

    float b = 0.0f;
    for (int i = tid; i < BCE_BLOCKS; i += 256) b += partial[i];

    float r = 0.0f;
    if (tid < N_GROUPS) {
        float gm = 0.0f;
#pragma unroll
        for (int j = 0; j < GROUP_W; ++j) gm += colnorm[tid * GROUP_W + j];
        gm *= (1.0f / GROUP_W);
        r += penalty_f(gm);
    }
    for (int i = tid; i < IN_FEATS - CAT_COLS; i += 256)
        r += penalty_f(colnorm[CAT_COLS + i]);

    sb[tid] = b;
    sr[tid] = r;
    __syncthreads();
    for (int s = 128; s > 0; s >>= 1) {
        if (tid < s) { sb[tid] += sb[tid + s]; sr[tid] += sr[tid + s]; }
        __syncthreads();
    }
    if (tid == 0) out[0] = sb[0] * (1.0f / (float)BATCH) + LAMBDA_C * sr[0];
}

// ---------------- Launch ----------------------------------------------------
extern "C" void kernel_launch(void* const* d_in, const int* in_sizes, int n_in,
                              void* d_out, int out_size, void* d_ws, size_t ws_size,
                              hipStream_t stream) {
    (void)in_sizes; (void)n_in; (void)out_size; (void)ws_size;
    const float* p = (const float*)d_in[0];
    const float* y = (const float*)d_in[1];
    const float* W = (const float*)d_in[2];
    float* out = (float*)d_out;
    float* ws  = (float*)d_ws;
    float* partial = ws;             // 2048 floats
    float* colnorm = ws + BCE_BLOCKS;// 2048 floats

    bce_partial_kernel<<<BCE_BLOCKS, BCE_THREADS, 0, stream>>>(p, y, partial);
    colnorm_wmma_kernel<<<16, 256, 0, stream>>>(W, colnorm);   // 128 waves = 128 col tiles
    final_combine_kernel<<<1, 256, 0, stream>>>(partial, colnorm, out);
}
